// MHAttention_48163763257429
// MI455X (gfx1250) — matrix operand order repaired and checked by reading, not compiled, not run on hardware
//
#include <hip/hip_runtime.h>

typedef __attribute__((ext_vector_type(16))) _Float16 v16h;
typedef __attribute__((ext_vector_type(8)))  _Float16 v8h;
typedef __attribute__((ext_vector_type(4)))  _Float16 v4h;
typedef __attribute__((ext_vector_type(8)))  float    v8f;
typedef __attribute__((ext_vector_type(4)))  float    v4f;

#define D_MODEL 1024
#define SEQ     2048
#define BATCH   2
#define NHEADS  16
#define HDIM    64
#define MROWS   (BATCH * SEQ)   // 4096

// ---------------------------------------------------------------------------
// x f32 -> f16 (vectorized, 4 elems/thread)
// ---------------------------------------------------------------------------
__global__ __launch_bounds__(256) void cvt_f32_f16(const float* __restrict__ X,
                                                   _Float16* __restrict__ Y) {
  const size_t i = (size_t)blockIdx.x * 256 + threadIdx.x;
  v4f v = ((const v4f*)X)[i];
  v4h h;
#pragma unroll
  for (int j = 0; j < 4; ++j) h[j] = (_Float16)v[j];
  ((v4h*)Y)[i] = h;
}

// ---------------------------------------------------------------------------
// Weight transpose + convert: W f32 [1024,1024] -> Wt f16 [1024,1024] (N-major)
// ---------------------------------------------------------------------------
__global__ __launch_bounds__(256) void transpose_w(const float* __restrict__ W0,
                                                   const float* __restrict__ W1,
                                                   const float* __restrict__ W2,
                                                   const float* __restrict__ W3,
                                                   _Float16* __restrict__ T0,
                                                   _Float16* __restrict__ T1,
                                                   _Float16* __restrict__ T2,
                                                   _Float16* __restrict__ T3) {
  __shared__ _Float16 tile[32][33];
  const float* W = (blockIdx.z == 0) ? W0 : (blockIdx.z == 1) ? W1
                   : (blockIdx.z == 2) ? W2 : W3;
  _Float16* T = (blockIdx.z == 0) ? T0 : (blockIdx.z == 1) ? T1
                : (blockIdx.z == 2) ? T2 : T3;
  const int tx = threadIdx.x, ty = threadIdx.y;
  const int r0 = blockIdx.y * 32, c0 = blockIdx.x * 32;
#pragma unroll
  for (int i = 0; i < 4; ++i)
    tile[ty + 8 * i][tx] = (_Float16)W[(size_t)(r0 + ty + 8 * i) * D_MODEL + c0 + tx];
  __syncthreads();
#pragma unroll
  for (int i = 0; i < 4; ++i)
    T[(size_t)(c0 + ty + 8 * i) * D_MODEL + r0 + tx] = tile[tx][ty + 8 * i];
}

// ---------------------------------------------------------------------------
// V transpose (per batch): Vh f16 [b][s][c] -> Vt f16 [b][c][s]
// ---------------------------------------------------------------------------
__global__ __launch_bounds__(256) void transpose_v(const _Float16* __restrict__ Vh,
                                                   _Float16* __restrict__ Vt) {
  __shared__ _Float16 tile[32][33];
  const int tx = threadIdx.x, ty = threadIdx.y;
  const int b = blockIdx.z;
  const int s0 = blockIdx.y * 32, c0 = blockIdx.x * 32;
  const _Float16* src = Vh + (size_t)b * SEQ * D_MODEL;
  _Float16* dst = Vt + (size_t)b * D_MODEL * SEQ;
#pragma unroll
  for (int i = 0; i < 4; ++i)
    tile[ty + 8 * i][tx] = src[(size_t)(s0 + ty + 8 * i) * D_MODEL + c0 + tx];
  __syncthreads();
#pragma unroll
  for (int i = 0; i < 4; ++i)
    dst[(size_t)(c0 + ty + 8 * i) * SEQ + s0 + tx] = tile[tx][ty + 8 * i];
}

// ---------------------------------------------------------------------------
// GEMM: A f16 [M,1024] @ Bt f16 [1024(N),1024(K)] -> C [M,1024]
// OUTMODE 0: f16 output; OUTMODE 1: f32 output.
// 256 thr (8 waves); wave tile 32x64; block tile 64x256; grid (M/64, N/256).
// B block-tile staged via GLOBAL_LOAD_ASYNC_TO_LDS, double-buffered,
// synced with s_wait_asynccnt + workgroup barrier. A frags direct b128 loads.
// ---------------------------------------------------------------------------
#define BROW_PITCH 40   // halves; 80B row pitch: 16B aligned, conflict-free reads

template <int OUTMODE>
__global__ __launch_bounds__(256) void gemm_h(const _Float16* __restrict__ A,
                                              const _Float16* __restrict__ Bt,
                                              void* __restrict__ Cout) {
  constexpr int K = D_MODEL, N = D_MODEL;
  __shared__ __align__(16) _Float16 ldsB[2][256][BROW_PITCH];  // 2 x 20KB
  const int tid = threadIdx.x;
  const int lane = tid & 31;
  const int wave = tid >> 5;
  const int wr = wave & 1, wc = wave >> 1;
  const int row0 = blockIdx.x * 64 + wr * 32;
  const int colB = blockIdx.y * 256;
  const int col0 = colB + wc * 64;
  const int m = lane & 15;
  const bool hiL = lane >= 16;
  const int kbA = hiL ? 8 : 0;    // unused; A frags load at kbB too
  const int kbB = hiL ? 16 : 0;   // B-frag K base (sequential halves)

  const _Float16* arow[2];
#pragma unroll
  for (int i = 0; i < 2; ++i) arow[i] = A + (size_t)(row0 + i * 16 + m) * K;
  const _Float16* bsrc = Bt + (size_t)(colB + tid) * K;   // thread owns one row

  auto stage = [&](int kk, int buf) {
    const unsigned l0 = (unsigned)(uintptr_t)&ldsB[buf][tid][0];
    const unsigned long long g0 = (unsigned long long)(uintptr_t)(bsrc + kk);
#pragma unroll
    for (int q = 0; q < 4; ++q) {
      const unsigned loff = l0 + q * 16;
      const unsigned long long ga = g0 + q * 16;
      asm volatile("global_load_async_to_lds_b128 %0, %1, off"
                   :: "v"(loff), "v"(ga) : "memory");
    }
  };

  stage(0, 0);
  v8f acc[2][4] = {};
  for (int kk = 0; kk < K; kk += 32) {
    const int buf = (kk >> 5) & 1;
    asm volatile("s_wait_asynccnt 0x0" ::: "memory");
    __syncthreads();                       // tile kk in LDS, prev buffer drained
    if (kk + 32 < K) stage(kk + 32, buf ^ 1);

    v16h a[2], bb[4];
#pragma unroll
    for (int i = 0; i < 2; ++i) {
      v8h p0 = *(const v8h*)(arow[i] + kk + kbB);
      v8h p1 = *(const v8h*)(arow[i] + kk + kbB + 8);
#pragma unroll
      for (int j = 0; j < 8; ++j) { a[i][j] = p0[j]; a[i][8 + j] = p1[j]; }
      __builtin_prefetch(arow[i] + kk + 256, 0, 1);
    }
#pragma unroll
    for (int jj = 0; jj < 4; ++jj) {
      const _Float16* lb = &ldsB[buf][wc * 64 + jj * 16 + m][0];
      v8h q0 = *(const v8h*)(lb + kbB);
      v8h q1 = *(const v8h*)(lb + kbB + 8);
#pragma unroll
      for (int j = 0; j < 8; ++j) { bb[jj][j] = q0[j]; bb[jj][8 + j] = q1[j]; }
    }
#pragma unroll
    for (int i = 0; i < 2; ++i)
#pragma unroll
      for (int jj = 0; jj < 4; ++jj)
        acc[i][jj] = __builtin_amdgcn_wmma_f32_16x16x32_f16(
            false, a[i], false, bb[jj], (short)0, acc[i][jj], false, false);
  }
#pragma unroll
  for (int i = 0; i < 2; ++i)
#pragma unroll
    for (int jj = 0; jj < 4; ++jj)
#pragma unroll
      for (int g = 0; g < 8; ++g) {
        const int r = row0 + i * 16 + g + (hiL ? 8 : 0);
        const int c = col0 + jj * 16 + m;
        if (OUTMODE == 0)
          ((_Float16*)Cout)[(size_t)r * N + c] = (_Float16)acc[i][jj][g];
        else
          ((float*)Cout)[(size_t)r * N + c] = acc[i][jj][g];
      }
}

// ---------------------------------------------------------------------------
// One 32-key attention step; MASK only instantiated for the diagonal tile.
// Row sums of P are computed with a WMMA against an all-ones B matrix
// (one matrix op replaces 32 ds_bpermute rounds).
// ---------------------------------------------------------------------------
template <bool MASK>
__device__ __forceinline__ void attn_tile(
    const _Float16* __restrict__ Km, const _Float16* __restrict__ Vt,
    _Float16 (&ldsP)[16][32], const v16h (&aq)[2], const v16h& ones,
    v8f (&o)[4], float (&rmax)[8], float (&rsum)[8],
    size_t baseRow, int b, int h, int qt, int key0,
    int m, bool hiL, int kbA, int kbB) {
  // --- S = Q.K^T for keys key0..key0+31
  v8f s0 = {}, s1 = {};
  const _Float16* krow0 = Km + (baseRow + key0 + m) * D_MODEL + h * HDIM;
  const _Float16* krow1 = krow0 + 16 * D_MODEL;
#pragma unroll
  for (int d2 = 0; d2 < 2; ++d2) {
    v16h bk0, bk1;
    v8h a0 = *(const v8h*)(krow0 + d2 * 32 + kbB);
    v8h a1 = *(const v8h*)(krow0 + d2 * 32 + kbB + 8);
    v8h b0 = *(const v8h*)(krow1 + d2 * 32 + kbB);
    v8h b1 = *(const v8h*)(krow1 + d2 * 32 + kbB + 8);
#pragma unroll
    for (int j = 0; j < 8; ++j) {
      bk0[j] = a0[j]; bk0[8 + j] = a1[j];
      bk1[j] = b0[j]; bk1[8 + j] = b1[j];
    }
    s0 = __builtin_amdgcn_wmma_f32_16x16x32_f16(false, aq[d2], false, bk0,
                                                (short)0, s0, false, false);
    s1 = __builtin_amdgcn_wmma_f32_16x16x32_f16(false, aq[d2], false, bk1,
                                                (short)0, s1, false, false);
  }

  // --- online softmax (max via half-wave shuffles; sum deferred to WMMA)
#pragma unroll
  for (int g = 0; g < 8; ++g) {
    const int rowm = g + (hiL ? 8 : 0);
    float v0 = s0[g] * 0.125f;                    // 1/sqrt(64)
    float v1 = s1[g] * 0.125f;
    if (MASK) {
      const int qpos = qt * 16 + rowm;
      if (key0 + m > qpos)      v0 = -3.0e38f;
      if (key0 + 16 + m > qpos) v1 = -3.0e38f;
    }
    float t = fmaxf(v0, v1);
#pragma unroll
    for (int off = 1; off < 16; off <<= 1) t = fmaxf(t, __shfl_xor(t, off, 32));
    const float nm = fmaxf(rmax[g], t);
    const float scale = __expf(rmax[g] - nm);
    rmax[g] = nm;
    rsum[g] *= scale;
#pragma unroll
    for (int t4 = 0; t4 < 4; ++t4) o[t4][g] = o[t4][g] * scale;
    ldsP[rowm][m]      = (_Float16)__expf(v0 - nm);
    ldsP[rowm][16 + m] = (_Float16)__expf(v1 - nm);
  }
  asm volatile("s_wait_dscnt 0" ::: "memory");    // wave-local LDS RAW

  // --- P as full 16x32 A-fragment
  v16h ap;
#pragma unroll
  for (int j = 0; j < 8; ++j) {
    ap[j]     = ldsP[m][kbB + j];
    ap[8 + j] = ldsP[m][kbB + 8 + j];
  }

  // --- row sums: psum = P x ones (all columns equal the row sum)
  v8f psum = {};
  psum = __builtin_amdgcn_wmma_f32_16x16x32_f16(false, ap, false, ones,
                                                (short)0, psum, false, false);
#pragma unroll
  for (int g = 0; g < 8; ++g) rsum[g] += psum[g];

  // --- O += P.V ; Vt rows contiguous in key
#pragma unroll
  for (int t4 = 0; t4 < 4; ++t4) {
    const _Float16* vrow =
        Vt + ((size_t)b * D_MODEL + h * HDIM + t4 * 16 + m) * SEQ + key0 + kbB;
    v8h q0 = *(const v8h*)(vrow);
    v8h q1 = *(const v8h*)(vrow + 8);
    v16h bv;
#pragma unroll
    for (int j = 0; j < 8; ++j) { bv[j] = q0[j]; bv[8 + j] = q1[j]; }
    o[t4] = __builtin_amdgcn_wmma_f32_16x16x32_f16(false, ap, false, bv,
                                                   (short)0, o[t4], false, false);
  }
}

// ---------------------------------------------------------------------------
// Causal flash attention: one wave per (batch, head, 16-row q-tile),
// 32 keys/iter; unmasked steady-state + masked diagonal epilogue.
// ---------------------------------------------------------------------------
__global__ __launch_bounds__(256) void attn_causal(const _Float16* __restrict__ Q,
                                                   const _Float16* __restrict__ Km,
                                                   const _Float16* __restrict__ Vt,
                                                   _Float16* __restrict__ ctx) {
  __shared__ _Float16 ldsP[8][16][32];   // per-wave 16x32 P tile
  const int lane = threadIdx.x & 31;
  const int wave = threadIdx.x >> 5;
  const int task = blockIdx.x * 8 + wave;   // 4096 = 2 * 16 * 128
  const int qt = task & 127;
  const int h  = (task >> 7) & 15;
  const int b  = task >> 11;
  const int m = lane & 15;
  const bool hiL = lane >= 16;
  const int kbA = hiL ? 8 : 0;
  const int kbB = hiL ? 16 : 0;
  const size_t baseRow = (size_t)b * SEQ;

  v16h aq[2];
  {
    const _Float16* qrow = Q + (baseRow + qt * 16 + m) * D_MODEL + h * HDIM;
#pragma unroll
    for (int d2 = 0; d2 < 2; ++d2) {
      v8h lo = *(const v8h*)(qrow + d2 * 32 + kbB);
      v8h hi = *(const v8h*)(qrow + d2 * 32 + kbB + 8);
#pragma unroll
      for (int j = 0; j < 8; ++j) { aq[d2][j] = lo[j]; aq[d2][8 + j] = hi[j]; }
    }
  }

  v16h ones;
#pragma unroll
  for (int j = 0; j < 16; ++j) ones[j] = (_Float16)1.0f;

  float rmax[8], rsum[8];
#pragma unroll
  for (int g = 0; g < 8; ++g) { rmax[g] = -3.0e38f; rsum[g] = 0.0f; }
  v8f o[4] = {};

  const int ntiles = qt / 2 + 1;   // 32-key tiles covering keys <= qt*16+15
  for (int kt = 0; kt < ntiles - 1; ++kt)
    attn_tile<false>(Km, Vt, ldsP[wave], aq, ones, o, rmax, rsum,
                     baseRow, b, h, qt, kt * 32, m, hiL, kbA, kbB);
  attn_tile<true>(Km, Vt, ldsP[wave], aq, ones, o, rmax, rsum,
                  baseRow, b, h, qt, (ntiles - 1) * 32, m, hiL, kbA, kbB);

#pragma unroll
  for (int t4 = 0; t4 < 4; ++t4)
#pragma unroll
    for (int g = 0; g < 8; ++g) {
      const int rowm = g + (hiL ? 8 : 0);
      const float val = o[t4][g] / rsum[g];
      ctx[(baseRow + qt * 16 + rowm) * D_MODEL + h * HDIM + t4 * 16 + m] = (_Float16)val;
    }
}

// ---------------------------------------------------------------------------
extern "C" void kernel_launch(void* const* d_in, const int* in_sizes, int n_in,
                              void* d_out, int out_size, void* d_ws, size_t ws_size,
                              hipStream_t stream) {
  (void)in_sizes; (void)n_in; (void)out_size; (void)ws_size;
  const float* x    = (const float*)d_in[0];
  const float* Wq   = (const float*)d_in[1];
  const float* Wk   = (const float*)d_in[2];
  const float* Wv   = (const float*)d_in[3];
  const float* Wout = (const float*)d_in[4];
  float* out = (float*)d_out;

  const size_t MNh = (size_t)MROWS * D_MODEL;   // 4M halves = 8MB
  const size_t WN  = (size_t)D_MODEL * D_MODEL; // 1M halves = 2MB
  _Float16* ws  = (_Float16*)d_ws;              // total 5*MNh halves = 40MB
  _Float16* xh  = ws;                 // later reused as Vt
  _Float16* WtQ = ws + MNh;
  _Float16* WtK = WtQ + WN;
  _Float16* WtV = WtK + WN;
  _Float16* WtO = WtV + WN;
  _Float16* Qh  = ws + 2 * MNh;
  _Float16* Kh  = ws + 3 * MNh;
  _Float16* Vh  = ws + 4 * MNh;       // later reused as ctx
  _Float16* Vtr = xh;                 // Vt overlays xh (xh dead after V proj)
  _Float16* Ch  = Vh;                 // ctx overlays Vh (Vh dead after transpose)

  dim3 blk(256);
  cvt_f32_f16<<<dim3((unsigned)(MNh / 4 / 256)), blk, 0, stream>>>(x, xh);
  transpose_w<<<dim3(32, 32, 4), dim3(32, 8), 0, stream>>>(Wq, Wk, Wv, Wout,
                                                           WtQ, WtK, WtV, WtO);
  dim3 gGemm(MROWS / 64, D_MODEL / 256);        // (64, 4)
  gemm_h<0><<<gGemm, blk, 0, stream>>>(xh, WtQ, (void*)Qh);
  gemm_h<0><<<gGemm, blk, 0, stream>>>(xh, WtK, (void*)Kh);
  gemm_h<0><<<gGemm, blk, 0, stream>>>(xh, WtV, (void*)Vh);
  transpose_v<<<dim3(D_MODEL / 32, SEQ / 32, BATCH), dim3(32, 8), 0, stream>>>(Vh, Vtr);
  attn_causal<<<dim3(BATCH * NHEADS * (SEQ / 16) / 8), blk, 0, stream>>>(Qh, Kh, Vtr, Ch);
  gemm_h<1><<<gGemm, blk, 0, stream>>>(Ch, WtO, (void*)out);
}
